// CNN_Mem_18846316495475
// MI455X (gfx1250) — matrix operand order repaired and checked by reading, not compiled, not run on hardware
//
#include <hip/hip_runtime.h>

// ---------------- dims ----------------
#define B_    128
#define L_    64
#define LP    68            // padded seq positions (64..67 zeroed)
#define D_    300
#define NF    300           // feature dim (3*100)
#define NFP   304           // padded feature cols
#define KC    1536          // conv GEMM K (5*300 padded to mult of 32)
#define TMAX  62            // conv output positions computed (t=0..61)
#define MROWS (B_*TMAX)     // 7936
#define MT_C  (MROWS/16)    // 496 M-tiles  (conv)
#define NT_C  (NFP/16)      // 19 N-tiles   (conv)
#define MEM   262144
#define KEY   300
#define KQ    320           // padded q K (300..319 zeroed)
#define NEG_BIG (-3.0e38f)

typedef __attribute__((ext_vector_type(16))) __bf16 v16bf;
typedef __attribute__((ext_vector_type(8)))  float  v8f;
typedef __attribute__((ext_vector_type(4)))  float  v4f;   // clang ext-vector (NT-load capable)

union Frag16 {
    v16bf v;
    __bf16             e[16];
    unsigned short     u[16];
    unsigned long long q[4];
    uint4              x[2];
};

__device__ __forceinline__ unsigned short f2bf(float f) {
    unsigned u = __float_as_uint(f);
    unsigned r = u + 0x7FFFu + ((u >> 16) & 1u);   // RNE
    return (unsigned short)(r >> 16);
}
__device__ __forceinline__ unsigned encf(float f) {   // monotonic float->uint
    unsigned u = __float_as_uint(f);
    return (u & 0x80000000u) ? ~u : (u | 0x80000000u);
}
__device__ __forceinline__ float decf(unsigned e) {
    unsigned u = (e & 0x80000000u) ? (e & 0x7FFFFFFFu) : ~e;
    return __uint_as_float(u);
}

// ---------------- workspace layout (bytes) ----------------
#define WS_E    0                                   // bf16 e  [B][68][300]   5,222,400
#define WS_WT   (WS_E  + (size_t)B_*LP*D_*2)        // bf16 Wt [304][1536]      933,888
#define WS_CO   (WS_WT + (size_t)NFP*KC*2)          // f32 convout [7936][304]
#define WS_Q    (WS_CO + (size_t)MROWS*NFP*4)       // bf16 q  [128][320]
#define WS_POS  (WS_Q  + (size_t)B_*KQ*2)           // u32 posbits [128]
#define WS_NEG  (WS_POS + 512)                      // u32 negbits [128]

// =====================================================================
// 1) gather embedding -> bf16, zero-pad positions 64..67
// =====================================================================
__global__ void k_embed(const int* __restrict__ x, const float* __restrict__ embed,
                        unsigned short* __restrict__ e) {
    int idx = blockIdx.x * 256 + threadIdx.x;        // B*68*300 = 2,611,200
    int d = idx % D_;
    int p = (idx / D_) % LP;
    int b = idx / (D_ * LP);
    float v = 0.0f;
    if (p < L_) {
        int tok = x[b * L_ + p];
        v = embed[(size_t)tok * D_ + d];
    }
    e[idx] = f2bf(v);
}

// =====================================================================
// 2) build transposed, zero-padded conv weights Wt[n][k], k = tap*300+d
// =====================================================================
__global__ void k_weights(const float* __restrict__ w3, const float* __restrict__ w4,
                          const float* __restrict__ w5, unsigned short* __restrict__ Wt) {
    int idx = blockIdx.x * 256 + threadIdx.x;        // 304*1536 = 466,944
    int n = idx / KC;
    int k = idx % KC;
    float v = 0.0f;
    if (n < NF && k < 5 * D_) {
        int tap = k / D_;
        int d   = k % D_;
        if (n < 100)      { if (tap < 3) v = w3[((n       * D_) + d) * 3 + tap]; }
        else if (n < 200) { if (tap < 4) v = w4[(((n-100) * D_) + d) * 4 + tap]; }
        else              { if (tap < 5) v = w5[(((n-200) * D_) + d) * 5 + tap]; }
    }
    Wt[idx] = f2bf(v);
}

// ---------- fragment helpers for the big GEMM ----------
// mem_keys is a 315 MB read-once stream: use non-temporal loads so it
// does not evict the hot 80 KB q working set from WGP$/L2.
__device__ __forceinline__ Frag16 load_a_frag(const float* __restrict__ row, int ka) {
    v4f f0 = __builtin_nontemporal_load((const v4f*)(row + ka));
    v4f f1 = __builtin_nontemporal_load((const v4f*)(row + ka + 4));
    v4f f2 = __builtin_nontemporal_load((const v4f*)(row + ka + 16));
    v4f f3 = __builtin_nontemporal_load((const v4f*)(row + ka + 20));
    Frag16 a;
    a.e[0]  = (__bf16)f0.x;  a.e[1]  = (__bf16)f0.y;  a.e[2]  = (__bf16)f0.z;  a.e[3]  = (__bf16)f0.w;
    a.e[4]  = (__bf16)f1.x;  a.e[5]  = (__bf16)f1.y;  a.e[6]  = (__bf16)f1.z;  a.e[7]  = (__bf16)f1.w;
    a.e[8]  = (__bf16)f2.x;  a.e[9]  = (__bf16)f2.y;  a.e[10] = (__bf16)f2.z;  a.e[11] = (__bf16)f2.w;
    a.e[12] = (__bf16)f3.x;  a.e[13] = (__bf16)f3.y;  a.e[14] = (__bf16)f3.z;  a.e[15] = (__bf16)f3.w;
    return a;
}
__device__ __forceinline__ Frag16 load_a_frag_tail(const float* __restrict__ row, int ka) {
    Frag16 a;
#pragma unroll
    for (int i = 0; i < 8; ++i) {
        int k0 = ka + i, k1 = ka + 16 + i;
        a.e[i]     = (k0 < KEY) ? (__bf16)__builtin_nontemporal_load(row + k0) : (__bf16)0.0f;
        a.e[8 + i] = (k1 < KEY) ? (__bf16)__builtin_nontemporal_load(row + k1) : (__bf16)0.0f;
    }
    return a;
}
__device__ __forceinline__ Frag16 load_b_frag(const unsigned short* __restrict__ qrow) {
    Frag16 b;
    b.x[0] = *(const uint4*)(qrow);
    b.x[1] = *(const uint4*)(qrow + 16);
    return b;
}

// =====================================================================
// 3) big GEMM: sims[128, 262144] via WMMA, mem rows on the M side.
//    One mem M-tile per wave (64 accumulator VGPRs). An empty asm
//    redefines the q pointer every tile iteration so LICM cannot hoist
//    the 80 loop-invariant B fragments (640 VGPRs!) and spill them to
//    scratch -- B frags stay as in-loop WGP$-hit global_load_b128s.
//    mem_keys (HBM stream) read exactly once with NT hints, f32 -> bf16
//    converted in-flight via packed hw cvt. Fused masked pos/neg max
//    epilogue -> atomicMax on monotonic uint encodings (deterministic).
// =====================================================================
__global__ void k_gemm(const unsigned short* __restrict__ q,
                       const float* __restrict__ mem_keys,
                       const int* __restrict__ mem_values,
                       const int* __restrict__ y,
                       unsigned* __restrict__ posb, unsigned* __restrict__ negb) {
    int wave = threadIdx.x >> 5;
    int lane = threadIdx.x & 31;
    int lo = lane & 15, hi = lane >> 4;

    int ycol[8];
#pragma unroll
    for (int j = 0; j < 8; ++j) ycol[j] = y[j * 16 + lo];   // lane's batch column per N-tile

    float posm[8], negm[8];
#pragma unroll
    for (int j = 0; j < 8; ++j) { posm[j] = NEG_BIG; negm[j] = NEG_BIG; }

    const v8f zero = {0.f,0.f,0.f,0.f,0.f,0.f,0.f,0.f};
    const unsigned short* qv = q;

    for (int tile = blockIdx.x * 8 + wave; tile < MEM / 16; tile += 4096) {  // 16384 tiles
        // opaque redefinition: makes q-fragment loads loop-variant (no LICM hoist+spill)
        asm volatile("" : "+s"(qv));

        long mbase = (long)tile * 16;
        const float* arow = mem_keys + (mbase + lo) * (long)KEY;

        v8f acc[8];
#pragma unroll
        for (int j = 0; j < 8; ++j) acc[j] = zero;

        // K steps 0..8 : fully in-bounds fast path
#pragma unroll
        for (int ks = 0; ks < 9; ++ks) {
            int ka = ks * 32 + hi * 8;
            Frag16 a = load_a_frag(arow, ka);
            int kb = ks * 32 + hi * 8;
#pragma unroll
            for (int j = 0; j < 8; ++j) {
                Frag16 bb = load_b_frag(qv + (size_t)(j * 16 + lo) * KQ + kb);
                acc[j] = __builtin_amdgcn_wmma_f32_16x16x32_bf16(
                            false, a.v, false, bb.v, (short)0, acc[j], false, false);
            }
        }
        // K step 9 : K=288..319, mask k>=300 (q side is zero-padded already)
        {
            int ka = 288 + hi * 8;
            Frag16 a = load_a_frag_tail(arow, ka);
            int kb = 288 + hi * 8;
#pragma unroll
            for (int j = 0; j < 8; ++j) {
                Frag16 bb = load_b_frag(qv + (size_t)(j * 16 + lo) * KQ + kb);
                acc[j] = __builtin_amdgcn_wmma_f32_16x16x32_bf16(
                            false, a.v, false, bb.v, (short)0, acc[j], false, false);
            }
        }
        // fused masked-max epilogue (labels depend only on r/hi; batch col on lane)
        int lab[8];
#pragma unroll
        for (int r = 0; r < 8; ++r) lab[r] = mem_values[mbase + r + hi * 8];
#pragma unroll
        for (int j = 0; j < 8; ++j) {
            int yv = ycol[j];
#pragma unroll
            for (int r = 0; r < 8; ++r) {
                float s = acc[j][r];
                bool p = (lab[r] == yv);
                posm[j] = p ? fmaxf(posm[j], s) : posm[j];
                negm[j] = p ? negm[j] : fmaxf(negm[j], s);
            }
        }
    }
    // combine the two half-wave M-row groups, then global atomic max
#pragma unroll
    for (int j = 0; j < 8; ++j) {
        float pv = posm[j]; pv = fmaxf(pv, __shfl_xor(pv, 16));
        float nv = negm[j]; nv = fmaxf(nv, __shfl_xor(nv, 16));
        if (hi == 0) {
            atomicMax(&posb[j * 16 + lo], encf(pv));
            atomicMax(&negb[j * 16 + lo], encf(nv));
        }
    }
}

// =====================================================================
// 4) conv as implicit GEMM with WMMA bf16: [7936 x 1536] x [1536 x 304]
//    A rows = overlapping windows into e (contiguous!), B = Wt columns
// =====================================================================
__global__ void k_conv(const unsigned short* __restrict__ e,
                       const unsigned short* __restrict__ Wt,
                       float* __restrict__ convout) {
    int wid  = (blockIdx.x * 256 + threadIdx.x) >> 5;   // 0..9423 (496*19)
    int lane = threadIdx.x & 31;
    int lo = lane & 15, hi = lane >> 4;
    int nt = wid / MT_C;
    int mt = wid % MT_C;

    int m = mt * 16 + lo;
    int b = m / TMAX, t = m % TMAX;
    const unsigned short* arow = e  + ((size_t)(b * LP + t)) * D_;   // contiguous K
    const unsigned short* brow = Wt + (size_t)(nt * 16 + lo) * KC;

    v8f acc = {0.f,0.f,0.f,0.f,0.f,0.f,0.f,0.f};
#pragma unroll 4
    for (int ks = 0; ks < KC / 32; ++ks) {
        int ka = ks * 32 + hi * 8;
        Frag16 a;
        a.q[0] = *(const unsigned long long*)(arow + ka);
        a.q[1] = *(const unsigned long long*)(arow + ka + 4);
        a.q[2] = *(const unsigned long long*)(arow + ka + 16);
        a.q[3] = *(const unsigned long long*)(arow + ka + 20);
        int kb = ks * 32 + hi * 8;
        Frag16 bb;
        bb.x[0] = *(const uint4*)(brow + kb);
        bb.x[1] = *(const uint4*)(brow + kb + 16);
        acc = __builtin_amdgcn_wmma_f32_16x16x32_bf16(
                false, a.v, false, bb.v, (short)0, acc, false, false);
    }
    int ncol = nt * 16 + lo;
#pragma unroll
    for (int r = 0; r < 8; ++r) {
        int mrow = mt * 16 + r + hi * 8;
        convout[(size_t)mrow * NFP + ncol] = acc[r];
    }
}

// =====================================================================
// 5) max-pool over valid t (62/61/60 per conv) + bias + relu + normalize
//    -> q bf16 [128][320] (zero-padded K)
// =====================================================================
__global__ void k_feat(const float* __restrict__ convout,
                       const float* __restrict__ b3, const float* __restrict__ b4,
                       const float* __restrict__ b5, unsigned short* __restrict__ q) {
    __shared__ float red[256];
    __shared__ float feats[NF];
    int b = blockIdx.x, tid = threadIdx.x;
    float lsq = 0.0f;
    for (int n = tid; n < NF; n += 256) {
        int tl = (n < 100) ? 62 : ((n < 200) ? 61 : 60);
        const float* col = convout + (size_t)(b * TMAX) * NFP + n;
        float mx = NEG_BIG;
        for (int t = 0; t < tl; ++t) mx = fmaxf(mx, col[(size_t)t * NFP]);
        float bias = (n < 100) ? b3[n] : ((n < 200) ? b4[n - 100] : b5[n - 200]);
        float f = fmaxf(mx + bias, 0.0f);
        feats[n] = f;
        lsq += f * f;
    }
    red[tid] = lsq;
    __syncthreads();
    for (int s = 128; s > 0; s >>= 1) {
        if (tid < s) red[tid] += red[tid + s];
        __syncthreads();
    }
    float inv = 1.0f / fmaxf(sqrtf(red[0]), 1e-12f);
    for (int n = tid; n < KQ; n += 256) {
        float f = (n < NF) ? feats[n] * inv : 0.0f;
        q[(size_t)b * KQ + n] = f2bf(f);
    }
}

// =====================================================================
// 6) init running-max encodings (0u == most negative float)
// =====================================================================
__global__ void k_init(unsigned* __restrict__ posb, unsigned* __restrict__ negb) {
    int t = threadIdx.x;
    if (t < B_) { posb[t] = 0u; negb[t] = 0u; }
}

// =====================================================================
// 7) finalize: loss = mean relu(neg - pos + margin); acc = mean(pos > neg)
// =====================================================================
__global__ void k_final(const unsigned* __restrict__ posb, const unsigned* __restrict__ negb,
                        float* __restrict__ out) {
    __shared__ float sl[128];
    __shared__ float sa[128];
    int t = threadIdx.x;
    float pv = decf(posb[t]);
    float nv = decf(negb[t]);
    sl[t] = fmaxf(nv - pv + 0.1f, 0.0f);
    sa[t] = (pv > nv) ? 1.0f : 0.0f;
    __syncthreads();
    for (int s = 64; s > 0; s >>= 1) {
        if (t < s) { sl[t] += sl[t + s]; sa[t] += sa[t + s]; }
        __syncthreads();
    }
    if (t == 0) {
        out[0] = sl[0] / (float)B_;
        out[1] = sa[0] / (float)B_;
    }
}

// =====================================================================
extern "C" void kernel_launch(void* const* d_in, const int* in_sizes, int n_in,
                              void* d_out, int out_size, void* d_ws, size_t ws_size,
                              hipStream_t stream) {
    (void)in_sizes; (void)n_in; (void)out_size; (void)ws_size;
    const int*   x    = (const int*)  d_in[0];
    const int*   y    = (const int*)  d_in[1];
    const float* emb  = (const float*)d_in[2];
    const float* w3   = (const float*)d_in[3];
    const float* b3   = (const float*)d_in[4];
    const float* w4   = (const float*)d_in[5];
    const float* b4   = (const float*)d_in[6];
    const float* w5   = (const float*)d_in[7];
    const float* b5   = (const float*)d_in[8];
    const float* mk   = (const float*)d_in[9];
    const int*   mv   = (const int*)  d_in[10];
    float* out = (float*)d_out;

    char* ws = (char*)d_ws;
    unsigned short* e_bf  = (unsigned short*)(ws + WS_E);
    unsigned short* Wt    = (unsigned short*)(ws + WS_WT);
    float*          co    = (float*)         (ws + WS_CO);
    unsigned short* qb    = (unsigned short*)(ws + WS_Q);
    unsigned*       posb  = (unsigned*)      (ws + WS_POS);
    unsigned*       negb  = (unsigned*)      (ws + WS_NEG);

    k_embed  <<<(B_*LP*D_)/256, 256, 0, stream>>>(x, emb, e_bf);
    k_weights<<<(NFP*KC)/256,   256, 0, stream>>>(w3, w4, w5, Wt);
    k_conv   <<<(MT_C*NT_C)/8,  256, 0, stream>>>(e_bf, Wt, co);
    k_feat   <<<B_,             256, 0, stream>>>(co, b3, b4, b5, qb);
    k_init   <<<1,              128, 0, stream>>>(posb, negb);
    k_gemm   <<<512,            256, 0, stream>>>(qb, mk, mv, y, posb, negb);
    k_final  <<<1,              128, 0, stream>>>(posb, negb, out);
}
